// UnimpNet_44573170598395
// MI455X (gfx1250) — matrix operand order repaired and checked by reading, not compiled, not run on hardware
//
#include <hip/hip_runtime.h>
#include <math.h>

// ---------------- problem constants ----------------
#define NN 40000      // nodes
#define EE 640000     // edges
#define FF 128        // features
#define HH 4          // heads
#define DD 32         // dim per head
#define HD 128        // H*D
#define VV 64         // vocab / output classes

#define APAD 136      // padded LDS row (halves): 272B = 68 dwords -> no bank conflicts

typedef __attribute__((ext_vector_type(16))) _Float16 v16h;
typedef __attribute__((ext_vector_type(8)))  float    v8f;

// ---------------------------------------------------------------------------
// Prep: weights -> f16, pre-swizzled into WMMA B-fragment order (contiguous k, as A uses) so each
// lane's 16 halves are contiguous (32B -> global_load_b128 x2 per fragment).
//
//   Wswz  : proj weights,  [colTile 0..31][kChunk 0..3][lane 0..31][i 0..15]
//           value = Wseg[(kChunk*32 + (lane>>4)*16 + i)][colTile*16 + (lane&15)]
//   Wlswz : classifier Wl, [colTile 0..3 ][kChunk 0..3][lane 0..31][i 0..15]
//   bqkvs : bias concat [512]
// ---------------------------------------------------------------------------
__global__ __launch_bounds__(256) void prep_weights(
    const float* __restrict__ Wq, const float* __restrict__ Wk,
    const float* __restrict__ Wv, const float* __restrict__ Ws,
    const float* __restrict__ Wl,
    const float* __restrict__ bq, const float* __restrict__ bk,
    const float* __restrict__ bv, const float* __restrict__ bs,
    _Float16* __restrict__ Wswz, _Float16* __restrict__ Wlswz,
    float* __restrict__ bqkvs)
{
    int t = blockIdx.x * 256 + threadIdx.x;
    if (t < 128 * 512) {                      // proj weight swizzle (65536)
        int i    = t & 15;
        int lane = (t >> 4) & 31;
        int kc   = (t >> 9) & 3;
        int ct   = t >> 11;                   // 0..31
        int col  = ct * 16 + (lane & 15);     // 0..511
        int kb   = kc * 32 + ((lane >> 4) << 4) + i;   // 0..127
        int seg = col >> 7, cc = col & 127;
        const float* W = (seg == 0) ? Wq : (seg == 1) ? Wk : (seg == 2) ? Wv : Ws;
        Wswz[t] = (_Float16)W[kb * 128 + cc];
    } else if (t < 128 * 512 + 128 * 64) {    // classifier weight swizzle (8192)
        int u = t - 128 * 512;
        int i    = u & 15;
        int lane = (u >> 4) & 31;
        int kc   = (u >> 9) & 3;
        int ct   = u >> 11;                   // 0..3
        int col  = ct * 16 + (lane & 15);     // 0..63
        int kb   = kc * 32 + ((lane >> 4) << 4) + i;
        Wlswz[u] = (_Float16)Wl[kb * 64 + col];
    } else if (t < 128 * 512 + 128 * 64 + 512) {
        int c = t - (128 * 512 + 128 * 64);
        int seg = c >> 7, cc = c & 127;
        const float* b = (seg == 0) ? bq : (seg == 1) ? bk : (seg == 2) ? bv : bs;
        bqkvs[c] = b[cc];
    }
}

// ---------------------------------------------------------------------------
// x_y = x + emb[y]  (stored f16 as WMMA A operand)
// ---------------------------------------------------------------------------
__global__ __launch_bounds__(256) void build_xh(
    const float* __restrict__ x, const int* __restrict__ y,
    const float* __restrict__ emb, _Float16* __restrict__ xh)
{
    int t = blockIdx.x * 256 + threadIdx.x;          // over N*F, exact
    int n = t >> 7, c = t & 127;
    int lab = y[n];
    xh[t] = (_Float16)(x[t] + emb[lab * FF + c]);
}

// ---------------------------------------------------------------------------
// Fused Q|K|V|S projection: [N x 128] @ [128 x 512] + bias.
// Per wave: one 16x16 f32 tile via 4x v_wmma_f32_16x16x32_f16.
// A tile in padded LDS (conflict-free ds_load_b128); B fragments are
// contiguous 32B lane loads from the pre-swizzled weight buffer.
// Grid: (N/16, 4), block 256 (8 waves = 128 output cols).
// ---------------------------------------------------------------------------
__global__ __launch_bounds__(256) void qkvs_gemm(
    const _Float16* __restrict__ xh, const _Float16* __restrict__ Wswz,
    const float* __restrict__ bqkvs,
    _Float16* __restrict__ qkvh,   // [N][384] = q|k|v (f16)
    float* __restrict__ skip)      // [N][128] = s-projection + bias (f32)
{
    __shared__ _Float16 aTile[16 * APAD];
    int wave = threadIdx.x >> 5;
    int lane = threadIdx.x & 31;
    int rowTile = blockIdx.x;                 // 0..2499
    int colTile = blockIdx.y * 8 + wave;      // 0..31

    const _Float16* aSrc = xh + (size_t)rowTile * 16 * 128;
    for (int i = threadIdx.x; i < 16 * 128; i += 256)
        aTile[(i >> 7) * APAD + (i & 127)] = aSrc[i];
    __syncthreads();

    int m = lane & 15;        // row within tile (A) / col within tile (B,D)
    int g = lane >> 4;        // lane half-group
    const v16h* Bw = (const v16h*)Wswz;       // fragment-ordered

    v8f acc = {};
#pragma unroll
    for (int kc = 0; kc < 4; ++kc) {
        int k0 = kc * 32;
        v16h a;
#pragma unroll
        for (int i = 0; i < 16; ++i) {
            int ka = k0 + (g << 4) + i;  // contiguous k layout (matches B swizzle)
            a[i] = aTile[m * APAD + ka];
        }
        v16h b = Bw[(colTile * 4 + kc) * 32 + lane];             // 32B contiguous
        acc = __builtin_amdgcn_wmma_f32_16x16x32_f16(
            false, a, false, b, (short)0, acc, false, false);
    }

    int n16 = colTile * 16 + m;               // global output column 0..511
    int seg = n16 >> 7;                       // 0=q 1=k 2=v 3=skip (uniform/wave)
    int cc  = n16 & 127;
    float bn = bqkvs[n16];
    if (seg < 3) {
#pragma unroll
        for (int r = 0; r < 8; ++r) {
            int row = rowTile * 16 + g * 8 + r;                  // ISA D layout
            qkvh[(size_t)row * 384 + seg * 128 + cc] = (_Float16)(acc[r] + bn);
        }
    } else {
#pragma unroll
        for (int r = 0; r < 8; ++r) {
            int row = rowTile * 16 + g * 8 + r;
            skip[(size_t)row * 128 + cc] = acc[r] + bn;
        }
    }
}

// ---------------------------------------------------------------------------
// Softmax state init
// ---------------------------------------------------------------------------
__global__ __launch_bounds__(256) void init_softmax(float* __restrict__ mmax,
                                                    float* __restrict__ denom)
{
    int t = blockIdx.x * 256 + threadIdx.x;       // over N*H, exact
    mmax[t]  = -INFINITY;
    denom[t] = 0.0f;
}

// ---------------------------------------------------------------------------
// Edge logits: alpha[e][h] = dot(q[dst], k[src]) / sqrt(D); atomic max to m.
// One wave32 per edge; lane handles dim (h*32 + lane) for all 4 heads.
// ---------------------------------------------------------------------------
__global__ __launch_bounds__(256) void edge_logits(
    const int* __restrict__ ei, const _Float16* __restrict__ qkvh,
    float* __restrict__ alpha, float* __restrict__ mmax)
{
    int wave = threadIdx.x >> 5;
    int lane = threadIdx.x & 31;
    int e = blockIdx.x * 8 + wave;
    if (e >= EE) return;
    int s = ei[e];
    int d = ei[EE + e];
    const _Float16* q = qkvh + (size_t)d * 384;          // q row
    const _Float16* k = qkvh + (size_t)s * 384 + 128;    // k row
    float p0 = (float)q[lane]      * (float)k[lane];
    float p1 = (float)q[32 + lane] * (float)k[32 + lane];
    float p2 = (float)q[64 + lane] * (float)k[64 + lane];
    float p3 = (float)q[96 + lane] * (float)k[96 + lane];
#pragma unroll
    for (int off = 16; off > 0; off >>= 1) {
        p0 += __shfl_xor(p0, off, 32);
        p1 += __shfl_xor(p1, off, 32);
        p2 += __shfl_xor(p2, off, 32);
        p3 += __shfl_xor(p3, off, 32);
    }
    if (lane < 4) {
        float val = (lane == 0) ? p0 : (lane == 1) ? p1 : (lane == 2) ? p2 : p3;
        val *= 0.17677669529663687f;                     // 1/sqrt(32)
        alpha[(size_t)e * HH + lane] = val;
        atomicMax(&mmax[(size_t)d * HH + lane], val);    // global_atomic_max_num_f32
    }
}

// ---------------------------------------------------------------------------
// ea = exp(alpha - m[dst]); denom[dst] += ea.  One thread per (e,h).
// ---------------------------------------------------------------------------
__global__ __launch_bounds__(256) void edge_softmax(
    const int* __restrict__ ei, float* __restrict__ alpha,
    const float* __restrict__ mmax, float* __restrict__ denom)
{
    int t = blockIdx.x * 256 + threadIdx.x;       // over E*H, exact
    int e = t >> 2, h = t & 3;
    int d = ei[EE + e];
    float ea = __expf(alpha[t] - mmax[(size_t)d * HH + h]);
    alpha[t] = ea;
    atomicAdd(&denom[(size_t)d * HH + h], ea);
}

// ---------------------------------------------------------------------------
// Weighted scatter: skip[dst] += (ea/denom[dst]) * v[src]. One wave per edge.
// ---------------------------------------------------------------------------
__global__ __launch_bounds__(256) void edge_scatter(
    const int* __restrict__ ei, const _Float16* __restrict__ qkvh,
    const float* __restrict__ alpha, const float* __restrict__ denom,
    float* __restrict__ skip)
{
    int wave = threadIdx.x >> 5;
    int lane = threadIdx.x & 31;
    int e = blockIdx.x * 8 + wave;
    if (e >= EE) return;
    int s = ei[e];
    int d = ei[EE + e];
    const _Float16* v = qkvh + (size_t)s * 384 + 256;    // v row
    float* dstRow = skip + (size_t)d * 128;
#pragma unroll
    for (int h = 0; h < HH; ++h) {
        float w = alpha[(size_t)e * HH + h] / denom[(size_t)d * HH + h];
        atomicAdd(&dstRow[h * 32 + lane], w * (float)v[h * 32 + lane]);
    }
}

// ---------------------------------------------------------------------------
// Classifier: [N x 128] @ [128 x 64] + bl.  Block: 32 rows x 64 cols, 8 waves.
// ---------------------------------------------------------------------------
__global__ __launch_bounds__(256) void classifier_gemm(
    const float* __restrict__ skip, const _Float16* __restrict__ Wlswz,
    const float* __restrict__ bl, float* __restrict__ logits)
{
    __shared__ _Float16 aTile[32 * APAD];
    int wave = threadIdx.x >> 5;
    int lane = threadIdx.x & 31;
    int rowBase = blockIdx.x * 32;

    for (int i = threadIdx.x; i < 32 * 128; i += 256)
        aTile[(i >> 7) * APAD + (i & 127)] =
            (_Float16)skip[(size_t)rowBase * 128 + i];
    __syncthreads();

    int rt = wave >> 2, ct = wave & 3;
    int m = lane & 15, g = lane >> 4;
    const v16h* Bw = (const v16h*)Wlswz;

    v8f acc = {};
#pragma unroll
    for (int kc = 0; kc < 4; ++kc) {
        int k0 = kc * 32;
        v16h a;
#pragma unroll
        for (int i = 0; i < 16; ++i) {
            int ka = k0 + (g << 4) + i;
            a[i] = aTile[(rt * 16 + m) * APAD + ka];
        }
        v16h b = Bw[(ct * 4 + kc) * 32 + lane];
        acc = __builtin_amdgcn_wmma_f32_16x16x32_f16(
            false, a, false, b, (short)0, acc, false, false);
    }
    int col = ct * 16 + m;                        // 0..63
    float bn = bl[col];
#pragma unroll
    for (int r = 0; r < 8; ++r) {
        int row = rowBase + rt * 16 + g * 8 + r;
        logits[(size_t)row * VV + col] = acc[r] + bn;
    }
}

// ---------------------------------------------------------------------------
extern "C" void kernel_launch(void* const* d_in, const int* in_sizes, int n_in,
                              void* d_out, int out_size, void* d_ws, size_t ws_size,
                              hipStream_t stream)
{
    (void)in_sizes; (void)n_in; (void)out_size; (void)ws_size;

    const float* x   = (const float*)d_in[0];
    const int*   y   = (const int*)  d_in[1];
    const int*   ei  = (const int*)  d_in[2];
    const float* emb = (const float*)d_in[3];
    const float* Wq  = (const float*)d_in[4];
    const float* bq  = (const float*)d_in[5];
    const float* Wk  = (const float*)d_in[6];
    const float* bk  = (const float*)d_in[7];
    const float* Wv  = (const float*)d_in[8];
    const float* bv  = (const float*)d_in[9];
    const float* Ws  = (const float*)d_in[10];
    const float* bs  = (const float*)d_in[11];
    const float* Wl  = (const float*)d_in[12];
    const float* bl  = (const float*)d_in[13];
    float* logits = (float*)d_out;

    // workspace carve (256B aligned)
    char* p = (char*)d_ws;
    auto carve = [&](size_t bytes) {
        void* r = (void*)p;
        p += (bytes + 255) & ~(size_t)255;
        return r;
    };
    _Float16* xh    = (_Float16*)carve((size_t)NN * FF * 2);   // 10.24 MB
    _Float16* Wswz  = (_Float16*)carve(128 * 512 * 2);
    _Float16* Wlswz = (_Float16*)carve(128 * 64 * 2);
    float*    bqkvs = (float*)   carve(512 * 4);
    _Float16* qkvh  = (_Float16*)carve((size_t)NN * 384 * 2);  // 30.72 MB
    float*    skip  = (float*)   carve((size_t)NN * 128 * 4);  // 20.48 MB
    float*    alpha = (float*)   carve((size_t)EE * HH * 4);   // 10.24 MB
    float*    mmax  = (float*)   carve((size_t)NN * HH * 4);
    float*    denom = (float*)   carve((size_t)NN * HH * 4);

    // 1) weight/bias prep (f32 -> f16, WMMA-fragment swizzle)
    prep_weights<<<(128 * 512 + 128 * 64 + 512 + 255) / 256, 256, 0, stream>>>(
        Wq, Wk, Wv, Ws, Wl, bq, bk, bv, bs, Wswz, Wlswz, bqkvs);

    // 2) x_y = x + emb[y] (f16)
    build_xh<<<(NN * FF) / 256, 256, 0, stream>>>(x, y, emb, xh);

    // 3) fused q|k|v|s projection (WMMA)
    qkvs_gemm<<<dim3(NN / 16, 4), 256, 0, stream>>>(xh, Wswz, bqkvs, qkvh, skip);

    // 4) softmax state init
    init_softmax<<<(NN * HH) / 256, 256, 0, stream>>>(mmax, denom);

    // 5) edge logits + segment max
    edge_logits<<<EE / 8, 256, 0, stream>>>(ei, qkvh, alpha, mmax);

    // 6) exp + segment sum
    edge_softmax<<<(EE * HH) / 256, 256, 0, stream>>>(ei, alpha, mmax, denom);

    // 7) weighted message scatter-add into skip accumulator
    edge_scatter<<<EE / 8, 256, 0, stream>>>(ei, qkvh, alpha, denom, skip);

    // 8) classifier (WMMA)
    classifier_gemm<<<NN / 32, 256, 0, stream>>>(skip, Wlswz, bl, logits);
}
